// MultiAttention_33749853012089
// MI455X (gfx1250) — compile-verified, matrix operand order checked by reading, not run on hardware
//
#include <hip/hip_runtime.h>

// ---------------------------------------------------------------------------
// MI455X (gfx1250) multi-context attention block, bf16 WMMA pipeline.
// All operands stored FRAGMENT-MAJOR (1KB blocks = 32 lanes x 32B) so WMMA
// fragments load with 2x b128 instead of 16x u16. Async global->LDS staging.
// ---------------------------------------------------------------------------

typedef __attribute__((ext_vector_type(16))) __bf16 v16bf;
typedef __attribute__((ext_vector_type(8)))  float  v8f;
typedef __attribute__((ext_vector_type(4)))  int    v4i;
typedef __bf16 bf16_t;

#define B_    32
#define IDF   256
#define Q_    1024
#define CDF   512
#define L_    128
#define NCTX  10
#define NIDF  (NCTX * IDF)   // 2560

#ifndef __has_builtin
#define __has_builtin(x) 0
#endif
#if __has_builtin(__builtin_amdgcn_global_load_async_to_lds_b128) && \
    __has_builtin(__builtin_amdgcn_s_wait_asynccnt)
#define HAVE_ASYNC_LDS 1
#else
#define HAVE_ASYNC_LDS 0
#endif

#if HAVE_ASYNC_LDS
typedef __attribute__((address_space(1))) v4i as1_v4i;   // global 128-bit chunk
typedef __attribute__((address_space(3))) v4i as3_v4i;   // LDS 128-bit chunk
__device__ __forceinline__ void async_copy16(void* lds, const void* g) {
  // per-lane 16B async copy, tracked by ASYNCcnt
  __builtin_amdgcn_global_load_async_to_lds_b128(
      (as1_v4i*)(unsigned long long)g,
      (as3_v4i*)(unsigned)(unsigned long long)lds, 0, 0);
}
#endif

// ---------------- fragment layout helpers (ISA 7.12.2) ----------------------
// A 16x32 tile: lane = m + 16*((k&8)>>3); element e: k as below. B tiles use
// the mirrored layout with lane = n. One 16x32 (or 32x16) tile = 512 bf16 =
// 1KB block, lane's 16 elements contiguous at block + lane*16.

__device__ __forceinline__ int lane_id() { return threadIdx.x & 31; }

// element index within a tile block for matrix element (kk in 0..31, idx=m|n)
__device__ __forceinline__ int frag_off(int kk, int idx) {
  const int lane = idx + ((kk & 8) << 1);
  const int e    = ((kk & 16) >> 1) + (kk & 6) + (kk & 1);
  return lane * 16 + e;
}

// inverse: k coordinate for (lane, e)
__device__ __forceinline__ int frag_k(int lane, int e) {
  return ((e & 8) << 1) + ((lane & 16) >> 1) + (e & 6) + (e & 1);
}

// fragment load: 32B contiguous per lane -> 2x b128
__device__ __forceinline__ v16bf load_frag(const bf16_t* block) {
  return *(const v16bf*)(block + lane_id() * 16);
}

__device__ __forceinline__ v8f wmma_bf16(v16bf a, v16bf b, v8f c) {
  return __builtin_amdgcn_wmma_f32_16x16x32_bf16(false, a, false, b,
                                                 (short)0, c, false, false);
}

// ---------------- packing kernels (fp32 row-major -> bf16 fragments) --------

// A fragments from [M][ld] f32 (trans: element (mg,kg) read at src[kg*ld+mg])
__global__ void pack_a_f32(const float* __restrict__ src, bf16_t* __restrict__ dst,
                           int KT, int ld, int trans) {
  const int idx  = blockIdx.x * blockDim.x + threadIdx.x; // one lane-fragment
  const int lane = idx & 31;
  const int tile = idx >> 5;
  const int kt   = tile % KT, mt = tile / KT;
  const int m    = lane & 15;
  v16bf f;
#pragma unroll
  for (int e = 0; e < 16; ++e) {
    const int kg = kt * 32 + frag_k(lane, e);
    const int mg = mt * 16 + m;
    f[e] = (__bf16)(trans ? src[(size_t)kg * ld + mg] : src[(size_t)mg * ld + kg]);
  }
  *(v16bf*)(dst + (size_t)tile * 512 + lane * 16) = f;
}

// B fragments from f32 (element (kg,ng); trans: read src[ng*ld+kg])
__global__ void pack_b_f32(const float* __restrict__ src, bf16_t* __restrict__ dst,
                           int NT, int ld, int trans) {
  const int idx  = blockIdx.x * blockDim.x + threadIdx.x;
  const int lane = idx & 31;
  const int tile = idx >> 5;
  const int nt   = tile % NT, kt = tile / NT;
  const int n    = lane & 15;
  v16bf f;
#pragma unroll
  for (int e = 0; e < 16; ++e) {
    const int kg = kt * 32 + frag_k(lane, e);
    const int ng = nt * 16 + n;
    f[e] = (__bf16)(trans ? src[(size_t)ng * ld + kg] : src[(size_t)kg * ld + ng]);
  }
  *(v16bf*)(dst + (size_t)tile * 512 + lane * 16) = f;
}

// targetT A-fragments: A[q][d] = input[b][d][q]; blocks ((b*64+m16)*8+kt)
__global__ void pack_tgt(const float* __restrict__ input, bf16_t* __restrict__ dst) {
  const int idx  = blockIdx.x * blockDim.x + threadIdx.x;
  const int lane = idx & 31;
  const int tile = idx >> 5;
  const int kt   = tile & 7;
  const int m16  = (tile >> 3) & 63;
  const int b    = tile >> 9;
  const int m    = lane & 15;
  v16bf f;
#pragma unroll
  for (int e = 0; e < 16; ++e) {
    const int d = kt * 32 + frag_k(lane, e);
    const int q = m16 * 16 + m;
    f[e] = (__bf16)input[((size_t)b * IDF + d) * Q_ + q];
  }
  *(v16bf*)(dst + (size_t)tile * 512 + lane * 16) = f;
}

// ---------------- Kernel 1: sourceT = bf16(Wc @ ctx) ------------------------
// 320 batches of [256x512]x[512x128]; WG = 64 rows x 128 cols, 8 waves.
// ctx K-slab packed to B-fragments in LDS; D scattered into BOTH fragment
// layouts of sourceT (QK B-operand and PV Bt-operand).

__global__ void proj_kernel(const float* __restrict__ ctx,     // [N*B][512][128]
                            const bf16_t* __restrict__ wcf,    // A-frags 16x32 tiles
                            bf16_t* __restrict__ srcB,         // B-frags [kt*8+nt]
                            bf16_t* __restrict__ srcBT) {      // Bt-frags [kt2*16+nt2]
  __shared__ __align__(32) bf16_t ctx_s[8 * 512];              // 8 n-tiles, 8KB
  const int batch = blockIdx.x >> 2;
  const int m0    = (blockIdx.x & 3) * 64;
  const int w     = threadIdx.x >> 5;
  const int mrow  = m0 + (w & 3) * 16;
  const int ntile = (w >> 2) * 4;
  const int lane  = lane_id();
  const int nl = lane & 15, hi = lane >> 4;

  const float* cbase = ctx + (size_t)batch * CDF * L_;

  v8f acc[4];
#pragma unroll
  for (int t = 0; t < 4; ++t) acc[t] = (v8f){};

  for (int k = 0; k < CDF; k += 32) {
    __syncthreads();
    { // pack one 32x128 K-slab into 8 B-fragment tiles (1 lane-frag/thread)
      const int nt = threadIdx.x >> 5, ln = threadIdx.x & 31, n = ln & 15;
      v16bf f;
#pragma unroll
      for (int e = 0; e < 16; ++e) {
        const int kk = frag_k(ln, e);
        f[e] = (__bf16)cbase[(size_t)(k + kk) * L_ + nt * 16 + n];
      }
      *(v16bf*)(ctx_s + (nt << 9) + ln * 16) = f;
    }
    __syncthreads();
    const v16bf a = load_frag(wcf + (size_t)(((mrow >> 4) * (CDF / 32)) + (k >> 5)) * 512);
#pragma unroll
    for (int t = 0; t < 4; ++t)
      acc[t] = wmma_bf16(a, load_frag(ctx_s + ((ntile + t) << 9)), acc[t]);
  }

  // scatter D into both fragment-major sourceT copies
  bf16_t* bB  = srcB  + (size_t)batch * IDF * L_;
  bf16_t* bBT = srcBT + (size_t)batch * IDF * L_;
#pragma unroll
  for (int t = 0; t < 4; ++t)
#pragma unroll
    for (int r = 0; r < 8; ++r) {
      const int mo = mrow + r + 8 * hi;            // 0..255 (output row, = K of QK)
      const int l  = (ntile + t) * 16 + nl;        // 0..127
      const __bf16 v = (__bf16)acc[t][r];
      bB [(size_t)(((mo >> 5) * 8  + (l  >> 4)) << 9) + frag_off(mo & 31, l  & 15)] = v;
      bBT[(size_t)(((l  >> 5) * 16 + (mo >> 4)) << 9) + frag_off(l  & 31, mo & 15)] = v;
    }
}

// ---------------- Kernel 2: scores -> softmax -> attn -> cat ----------------
// WG per (n, b, q-tile of 128); LDS: srcB 64KB + srcBT 64KB + P 32KB = 160KB.
// Staging is a straight byte copy of fragment-major tiles -> async b128.

__global__ void attn_kernel(const bf16_t* __restrict__ tgtf,   // A-frags
                            const bf16_t* __restrict__ srcB,
                            const bf16_t* __restrict__ srcBT,
                            bf16_t* __restrict__ catf) {       // A-frags for out GEMM
  extern __shared__ __align__(32) char smem_raw[];
  bf16_t* sB = (bf16_t*)smem_raw;                      // 64KB
  bf16_t* sT = (bf16_t*)(smem_raw + 65536);            // 64KB
  bf16_t* pS = (bf16_t*)(smem_raw + 131072);           // 32KB (A-frags of P)

  const int wg = blockIdx.x;
  const int qt = wg & 7;
  const int b  = (wg >> 3) & 31;
  const int nc = wg >> 8;
  const int w  = threadIdx.x >> 5;
  const int lane = lane_id();
  const int nl = lane & 15, hi = lane >> 4;
  const size_t tb = (size_t)(nc * B_ + b) * IDF * L_;

  { // stage both source-tile fragment images (pure copy)
    const char* gB = (const char*)(srcB + tb);
    const char* gT = (const char*)(srcBT + tb);
#if HAVE_ASYNC_LDS
    for (int i = threadIdx.x; i < 4096; i += 256) {
      async_copy16((char*)sB + i * 16, gB + i * 16);
      async_copy16((char*)sT + i * 16, gT + i * 16);
    }
    __builtin_amdgcn_s_wait_asynccnt(0);
#else
    for (int i = threadIdx.x; i < 4096; i += 256) {
      ((uint4*)sB)[i] = ((const uint4*)gB)[i];
      ((uint4*)sT)[i] = ((const uint4*)gT)[i];
    }
#endif
  }
  __syncthreads();

  // ---- scores = T(128x256) @ S(256x128); wave stripe 16x128 ----
  const bf16_t* abase = tgtf + (size_t)((b * 64 + qt * 8 + w) * 8) * 512;
  v8f sc[8];
#pragma unroll
  for (int t = 0; t < 8; ++t) sc[t] = (v8f){};
  for (int k = 0; k < IDF; k += 32) {
    const v16bf a = load_frag(abase + (size_t)(k >> 5) * 512);
#pragma unroll
    for (int t = 0; t < 8; ++t)
      sc[t] = wmma_bf16(a, load_frag(sB + (((k >> 5) * 8 + t) << 9)), sc[t]);
  }

  // ---- softmax over L (each 16-lane half holds one row per accumulator r) --
#pragma unroll
  for (int r = 0; r < 8; ++r) {
    float mx = -3.4e38f;
#pragma unroll
    for (int t = 0; t < 8; ++t) mx = fmaxf(mx, sc[t][r]);
    for (int off = 1; off < 16; off <<= 1) mx = fmaxf(mx, __shfl_xor(mx, off, 32));
    float s = 0.f;
#pragma unroll
    for (int t = 0; t < 8; ++t) { float e = __expf(sc[t][r] - mx); sc[t][r] = e; s += e; }
    for (int off = 1; off < 16; off <<= 1) s += __shfl_xor(s, off, 32);
    const float inv = 1.f / s;
#pragma unroll
    for (int t = 0; t < 8; ++t) sc[t][r] *= inv;
  }

  // P -> LDS as A-fragments: blocks (w*4 + kt2)
#pragma unroll
  for (int t = 0; t < 8; ++t)
#pragma unroll
    for (int r = 0; r < 8; ++r) {
      const int m = r + 8 * hi;
      const int l = t * 16 + nl;
      pS[((w * 4 + (l >> 5)) << 9) + frag_off(l & 31, m)] = (__bf16)sc[t][r];
    }
  __syncthreads();

  // ---- attn = P(128x128) @ S^T(128x256); 64-col chunks ----
  for (int dc = 0; dc < 4; ++dc) {
    v8f av[4];
#pragma unroll
    for (int t = 0; t < 4; ++t) av[t] = (v8f){};
    for (int k = 0; k < L_; k += 32) {
      const v16bf a = load_frag(pS + ((w * 4 + (k >> 5)) << 9));
#pragma unroll
      for (int t = 0; t < 4; ++t)
        av[t] = wmma_bf16(a, load_frag(sT + ((((k >> 5) * 16) + dc * 4 + t) << 9)), av[t]);
    }
    // scatter into cat A-fragments: blocks ((b*64 + q>>4)*80 + f>>5)
#pragma unroll
    for (int t = 0; t < 4; ++t)
#pragma unroll
      for (int r = 0; r < 8; ++r) {
        const int q = qt * 128 + w * 16 + r + 8 * hi;
        const int f = nc * IDF + dc * 64 + t * 16 + nl;
        catf[((size_t)((b * 64 + (q >> 4)) * 80 + (f >> 5)) << 9) +
             frag_off(f & 31, q & 15)] = (__bf16)av[t][r];
      }
  }
}

// ---------------- Kernel 3: x = Wcat@cat + b; relu; +res; LayerNorm ---------
// Single K-pass (K=2560): each cat fragment loaded once, amortized over all
// 16 output n-tiles (128 accumulator VGPRs/lane). Then LDS row-wise epilogue.

__global__ void out_kernel(const bf16_t* __restrict__ catf,   // A-frags
                           const bf16_t* __restrict__ wcatf,  // B-frags (kt*16+nt)
                           const float* __restrict__ input,
                           const float* __restrict__ wb,
                           const float* __restrict__ gamma,
                           const float* __restrict__ beta,
                           float* __restrict__ out) {
  extern __shared__ __align__(32) char smem_raw[];
  float* x_s = (float*)smem_raw;                              // [128][256] 128KB

  const int b  = blockIdx.x >> 3;
  const int qt = blockIdx.x & 7;
  const int w  = threadIdx.x >> 5;
  const int lane = lane_id();
  const int nl = lane & 15, hi = lane >> 4;

  const bf16_t* abase = catf + (size_t)((b * 64 + qt * 8 + w) * 80) * 512;

  v8f acc[16];
#pragma unroll
  for (int t = 0; t < 16; ++t) acc[t] = (v8f){};

#pragma unroll 1
  for (int kt = 0; kt < NIDF / 32; ++kt) {
    __builtin_prefetch(abase + (size_t)(kt + 4) * 512, 0, 0); // global_prefetch_b8
    const v16bf a = load_frag(abase + (size_t)kt * 512);
    const bf16_t* bbase = wcatf + (size_t)kt * 16 * 512;
#pragma unroll
    for (int t = 0; t < 16; ++t)
      acc[t] = wmma_bf16(a, load_frag(bbase + (size_t)t * 512), acc[t]);
  }

#pragma unroll
  for (int t = 0; t < 16; ++t)
#pragma unroll
    for (int r = 0; r < 8; ++r)
      x_s[(w * 16 + r + 8 * hi) * IDF + t * 16 + nl] = acc[t][r];
  __syncthreads();

  // row-wise epilogue: thread t < 128 owns row q0+t
  const int t = threadIdx.x;
  if (t < 128) {
    const int q = qt * 128 + t;
    float s = 0.f, s2 = 0.f;
    for (int d = 0; d < IDF; ++d) {
      float v = x_s[t * IDF + d] + wb[d];
      v = fmaxf(v, 0.f) + input[((size_t)b * IDF + d) * Q_ + q];
      x_s[t * IDF + d] = v;
      s += v; s2 += v * v;
    }
    const float mu   = s * (1.f / IDF);
    const float var  = s2 * (1.f / IDF) - mu * mu;
    const float rstd = rsqrtf(fmaxf(var, 0.f));
    for (int d = 0; d < IDF; ++d)
      out[((size_t)b * Q_ + q) * IDF + d] =
          (x_s[t * IDF + d] - mu) * rstd * gamma[d] + beta[d];
  }
}

// ---------------------------------------------------------------------------

extern "C" void kernel_launch(void* const* d_in, const int* in_sizes, int n_in,
                              void* d_out, int out_size, void* d_ws, size_t ws_size,
                              hipStream_t stream) {
  const float* input = (const float*)d_in[0];
  const float* ctx   = (const float*)d_in[1];
  const float* Wc    = (const float*)d_in[2];
  const float* Wcat  = (const float*)d_in[3];
  const float* Wb    = (const float*)d_in[4];
  const float* gamma = (const float*)d_in[5];
  const float* beta  = (const float*)d_in[6];
  float* out = (float*)d_out;

  // workspace layout (bf16 fragment-major intermediates)
  bf16_t* tgtf  = (bf16_t*)d_ws;                               //  16.8 MB
  bf16_t* wcf   = tgtf  + (size_t)B_ * Q_ * IDF;               //   0.25 MB
  bf16_t* wcatf = wcf   + (size_t)IDF * CDF;                   //   1.25 MB
  bf16_t* srcB  = wcatf + (size_t)IDF * NIDF;                  //  21 MB
  bf16_t* srcBT = srcB  + (size_t)NCTX * B_ * IDF * L_;        //  21 MB
  bf16_t* catf  = srcBT + (size_t)NCTX * B_ * IDF * L_;        // 167.8 MB

  (void)hipFuncSetAttribute((const void*)attn_kernel,
      hipFuncAttributeMaxDynamicSharedMemorySize, 160 * 1024);
  (void)hipFuncSetAttribute((const void*)out_kernel,
      hipFuncAttributeMaxDynamicSharedMemorySize, 128 * 1024);

  // 0) packing: weights + transposed target into fragment layouts
  pack_a_f32<<<(16 * 16 * 32) / 256, 256, 0, stream>>>(Wc, wcf, CDF / 32, CDF, 0);
  pack_b_f32<<<(80 * 16 * 32) / 256, 256, 0, stream>>>(Wcat, wcatf, IDF / 16, NIDF, 1);
  pack_tgt  <<<(B_ * 64 * 8 * 32) / 256, 256, 0, stream>>>(input, tgtf);

  // 1) context projection (WMMA), emits both sourceT fragment images
  proj_kernel<<<NCTX * B_ * 4, 256, 0, stream>>>(ctx, wcf, srcB, srcBT);

  // 2) attention per (n, b, q-tile): QK WMMA -> softmax -> PV WMMA
  attn_kernel<<<NCTX * B_ * (Q_ / 128), 256, 160 * 1024, stream>>>(
      tgtf, srcB, srcBT, catf);

  // 3) concat GEMM (K=2560) + bias + relu + residual + layernorm
  out_kernel<<<B_ * (Q_ / 128), 256, 128 * 1024, stream>>>(
      catf, wcatf, input, Wb, gamma, beta, out);
}
